// FunctionG_90632399880671
// MI455X (gfx1250) — matrix operand order repaired and checked by reading, not compiled, not run on hardware
//
#include <hip/hip_runtime.h>

typedef __attribute__((ext_vector_type(16))) _Float16 v16h;
typedef __attribute__((ext_vector_type(8)))  _Float16 v8h;
typedef __attribute__((ext_vector_type(8)))  float    v8f;

#define EMBD 224
#define HID  256
#define G3   768   // 3*HID

__device__ __forceinline__ float sigmoidf_(float x) { return 1.0f / (1.0f + __expf(-x)); }

// ---------------------------------------------------------------- f32 -> f16
__global__ void cvt_f16_kernel(const float* __restrict__ src, _Float16* __restrict__ dst, int n) {
  int i = blockIdx.x * blockDim.x + threadIdx.x;
  if (i < n) dst[i] = (_Float16)src[i];
}

// ------------------------------------------- gather embeddings for (B,T,3) trace -> (B*T,224) f16
__global__ void gather_trace_kernel(const int* __restrict__ trace, const int* __restrict__ uid_p,
                                    const float* __restrict__ uid_emb, const float* __restrict__ loc_emb,
                                    const float* __restrict__ wd_emb, const float* __restrict__ hr_emb,
                                    _Float16* __restrict__ out, int total) {
  int idx = blockIdx.x * blockDim.x + threadIdx.x;
  if (idx >= total) return;
  int col = idx % EMBD;
  int row = idx / EMBD;            // row = b*T + t  (matches trace layout)
  const int* p = trace + row * 3;
  float v;
  if (col < 64)       v = uid_emb[uid_p[0] * 64 + col];
  else if (col < 192) v = loc_emb[(size_t)p[0] * 128 + (col - 64)];
  else if (col < 208) v = wd_emb[p[1] * 16 + (col - 192)];
  else                v = hr_emb[p[2] * 16 + (col - 208)];
  out[idx] = (_Float16)v;
}

// ---------------------- gather candidate last-point embeddings -> (4096,224) f16
__global__ void gather_cand_kernel(const int* __restrict__ cand, const int* __restrict__ cur_trace,
                                   const int* __restrict__ uid_p,
                                   const float* __restrict__ uid_emb, const float* __restrict__ loc_emb,
                                   const float* __restrict__ wd_emb, const float* __restrict__ hr_emb,
                                   _Float16* __restrict__ out, int total) {
  int idx = blockIdx.x * blockDim.x + threadIdx.x;
  if (idx >= total) return;
  int col = idx % EMBD;
  int row = idx / EMBD;            // row = s*128 + c
  int s = row >> 7;
  float v;
  if (col < 64)       v = uid_emb[uid_p[0] * 64 + col];
  else if (col < 192) v = loc_emb[(size_t)cand[row] * 128 + (col - 64)];
  else if (col < 208) v = wd_emb[cur_trace[(s * 32 + 31) * 3 + 1] * 16 + (col - 192)];
  else                v = hr_emb[cur_trace[(s * 32 + 31) * 3 + 2] * 16 + (col - 208)];
  out[idx] = (_Float16)v;
}

// ---------------------------------------------------------------- WMMA GEMM
// D[M,N] = act( A[M,K](f16,lda) @ W[N,K](f16)^T + bias[N] ), one 16x16 tile per wave.
__global__ void gemm_xwT_kernel(const _Float16* __restrict__ A, long lda,
                                const _Float16* __restrict__ W,
                                const float* __restrict__ bias, int act,
                                float* __restrict__ D, long ldd,
                                int M, int N, int K) {
  int gw = (blockIdx.x * blockDim.x + threadIdx.x) >> 5;
  int ntn = N >> 4;
  if (gw >= (M >> 4) * ntn) return;
  int tm = gw / ntn, tn = gw % ntn;
  int lane = threadIdx.x & 31, half = lane >> 4, mrow = lane & 15;
  const _Float16* arow = A + (size_t)(tm * 16 + mrow) * lda;
  const _Float16* brow = W + (size_t)(tn * 16 + mrow) * K;
  v8f acc = {};
  for (int kb = 0; kb < K; kb += 32) {
    v16h af = *(const v16h*)(arow + kb + half * 16);
    v16h bf = *(const v16h*)(brow + kb + half * 16);
    acc = __builtin_amdgcn_wmma_f32_16x16x32_f16(false, af, false, bf, (short)0, acc, false, false);
  }
  int coln = tn * 16 + mrow;
  float bb = bias ? bias[coln] : 0.0f;
#pragma unroll
  for (int j = 0; j < 8; ++j) {
    float v = acc[j] + bb;
    if (act) v = fmaxf(v, 0.0f);
    D[(size_t)(tm * 16 + half * 8 + j) * ldd + coln] = v;
  }
}

// --------------------------------------- persistent GRU recurrence, batch 32, one workgroup
// h kept in LDS (f32 state + f16 copy for WMMA A-fragments). gh = h @ Whh^T staged in global scratch.
__global__ void __launch_bounds__(1024) gru_seq_kernel(const float* __restrict__ xproj,   // (32,T,768)
                                                       const _Float16* __restrict__ Whh16, // (768,256)
                                                       const float* __restrict__ b_hh,
                                                       float* __restrict__ gh,             // (32,768) scratch
                                                       float* __restrict__ hs,             // (32,T,256)
                                                       _Float16* __restrict__ hs16, int T) {
  __shared__ float    h32[32 * HID];
  __shared__ _Float16 h16[32 * HID];
  const int tid = threadIdx.x, wid = tid >> 5, lane = tid & 31;
  const int half = lane >> 4, mrow = lane & 15;
  for (int i = tid; i < 32 * HID; i += 1024) { h32[i] = 0.0f; h16[i] = (_Float16)0.0f; }
  __syncthreads();
  for (int t = 0; t < T; ++t) {
    // gh = h @ Whh^T + b_hh : 2x48 = 96 wmma tiles over 32 waves
    for (int tile = wid; tile < 96; tile += 32) {
      int tm = tile / 48, tn = tile % 48;
      const _Float16* arow = h16 + (tm * 16 + mrow) * HID;                 // LDS -> ds_load
      const _Float16* brow = Whh16 + (size_t)(tn * 16 + mrow) * HID;
      v8f acc = {};
#pragma unroll
      for (int kb = 0; kb < HID; kb += 32) {
        v16h af = *(const v16h*)(arow + kb + half * 16);
        v16h bf = *(const v16h*)(brow + kb + half * 16);
        acc = __builtin_amdgcn_wmma_f32_16x16x32_f16(false, af, false, bf, (short)0, acc, false, false);
      }
      int coln = tn * 16 + mrow;
      float bb = b_hh[coln];
#pragma unroll
      for (int j = 0; j < 8; ++j) gh[(tm * 16 + half * 8 + j) * G3 + coln] = acc[j] + bb;
    }
    __syncthreads();
    // gate combine: h' = (1-z)*n + z*h
    for (int i = tid; i < 32 * HID; i += 1024) {
      int b = i >> 8, hh = i & 255;
      const float* xp = xproj + ((size_t)b * T + t) * G3;
      const float* g = gh + b * G3;
      float r = sigmoidf_(xp[hh] + g[hh]);
      float z = sigmoidf_(xp[HID + hh] + g[HID + hh]);
      float n = tanhf(xp[2 * HID + hh] + r * g[2 * HID + hh]);
      float hv = (1.0f - z) * n + z * h32[i];
      h32[i] = hv;
      h16[i] = (_Float16)hv;
      hs[((size_t)b * T + t) * HID + hh] = hv;
      hs16[((size_t)b * T + t) * HID + hh] = (_Float16)hv;
    }
    __syncthreads();
  }
}

// ------------------------- final GRU step for all 4096 candidates (gh depends only on s)
__global__ void cand_gru_combine_kernel(const float* __restrict__ xp, const float* __restrict__ gh_s,
                                        const float* __restrict__ cur_hs,
                                        float* __restrict__ cand_h, _Float16* __restrict__ cand_h16) {
  int i = blockIdx.x * blockDim.x + threadIdx.x;
  if (i >= 4096 * HID) return;
  int row = i >> 8, hh = i & 255, s = row >> 7;
  const float* x = xp + (size_t)row * G3;
  const float* g = gh_s + s * G3;
  float hp = cur_hs[(s * 32 + 31) * HID + hh];
  float r = sigmoidf_(x[hh] + g[hh]);
  float z = sigmoidf_(x[HID + hh] + g[HID + hh]);
  float n = tanhf(x[2 * HID + hh] + r * g[2 * HID + hh]);
  float hv = (1.0f - z) * n + z * hp;
  cand_h[i] = hv;
  cand_h16[i] = (_Float16)hv;
}

// ------------------------------ intra attention + context (history encode), one block per sequence
__global__ void attn_ctx_kernel(const float* __restrict__ q, const float* __restrict__ k,
                                const float* __restrict__ hs, const float* __restrict__ outw,
                                float* __restrict__ ctx, _Float16* __restrict__ ctx16, int T) {
  int b = blockIdx.x, tid = threadIdx.x, wid = tid >> 5, lane = tid & 31;
  __shared__ float a[64], w[64];
  const float* qb = q + b * HID;
  for (int t = wid; t < T; t += 8) {
    const float* kt = k + ((size_t)b * T + t) * HID;
    float s = 0.0f;
#pragma unroll
    for (int j = 0; j < 8; ++j) { int h = lane + 32 * j; s += tanhf(qb[h] + kt[h]) * outw[h]; }
#pragma unroll
    for (int o = 16; o > 0; o >>= 1) s += __shfl_xor(s, o, 32);
    if (lane == 0) a[t] = s;
  }
  __syncthreads();
  if (tid == 0) {
    float m = -1e30f;
    for (int t = 0; t < T; ++t) m = fmaxf(m, a[t]);
    float ss = 0.0f;
    for (int t = 0; t < T; ++t) { float e = __expf(a[t] - m); w[t] = e; ss += e; }
    float inv = 1.0f / ss;
    for (int t = 0; t < T; ++t) w[t] *= inv;
  }
  __syncthreads();
  int h = tid;
  float c = 0.0f;
  for (int t = 0; t < T; ++t) c += w[t] * hs[((size_t)b * T + t) * HID + h];
  ctx[b * HID + h] = c;
  ctx16[b * HID + h] = (_Float16)c;
}

// --------- intra attention for candidates (shared 32-step prefix + 1 candidate step), block per (s,c)
__global__ void cand_attn_kernel(const float* __restrict__ q, const float* __restrict__ k_cur,
                                 const float* __restrict__ k_last, const float* __restrict__ cur_hs,
                                 const float* __restrict__ cand_h, const float* __restrict__ outw,
                                 float* __restrict__ ctx, _Float16* __restrict__ ctx16) {
  int i = blockIdx.x, s = i >> 7, tid = threadIdx.x, wid = tid >> 5, lane = tid & 31;
  __shared__ float a[33], w[33];
  const float* qb = q + (size_t)i * HID;
  for (int t = wid; t < 33; t += 8) {
    const float* kt = (t < 32) ? (k_cur + (size_t)(s * 32 + t) * HID) : (k_last + (size_t)i * HID);
    float sv = 0.0f;
#pragma unroll
    for (int j = 0; j < 8; ++j) { int h = lane + 32 * j; sv += tanhf(qb[h] + kt[h]) * outw[h]; }
#pragma unroll
    for (int o = 16; o > 0; o >>= 1) sv += __shfl_xor(sv, o, 32);
    if (lane == 0) a[t] = sv;
  }
  __syncthreads();
  if (tid == 0) {
    float m = -1e30f;
    for (int t = 0; t < 33; ++t) m = fmaxf(m, a[t]);
    float ss = 0.0f;
    for (int t = 0; t < 33; ++t) { float e = __expf(a[t] - m); w[t] = e; ss += e; }
    float inv = 1.0f / ss;
    for (int t = 0; t < 33; ++t) w[t] *= inv;
  }
  __syncthreads();
  int h = tid;
  float c = w[32] * cand_h[(size_t)i * HID + h];
  for (int t = 0; t < 32; ++t) c += w[t] * cur_hs[(size_t)(s * 32 + t) * HID + h];
  ctx[(size_t)i * HID + h] = c;
  ctx16[(size_t)i * HID + h] = (_Float16)c;
}

// --------- inter attention + moving + score, one block per candidate row i
__global__ void inter_attn_kernel(const float* __restrict__ qh2, const float* __restrict__ kh2,
                                  const float* __restrict__ outw, const float* __restrict__ hist_hidden,
                                  const float* __restrict__ score_w,
                                  float* __restrict__ moving, float* __restrict__ scores) {
  int i = blockIdx.x, tid = threadIdx.x, wid = tid >> 5, lane = tid & 31;
  __shared__ float a[32], w[32], mv[HID];
  const float* qb = qh2 + (size_t)i * HID;
  for (int j = wid; j < 32; j += 8) {
    const float* kj = kh2 + j * HID;
    float sv = 0.0f;
#pragma unroll
    for (int jj = 0; jj < 8; ++jj) { int h = lane + 32 * jj; sv += tanhf(qb[h] + kj[h]) * outw[h]; }
#pragma unroll
    for (int o = 16; o > 0; o >>= 1) sv += __shfl_xor(sv, o, 32);
    if (lane == 0) a[j] = sv;
  }
  __syncthreads();
  if (tid == 0) {
    float m = -1e30f;
    for (int j = 0; j < 32; ++j) m = fmaxf(m, a[j]);
    float ss = 0.0f;
    for (int j = 0; j < 32; ++j) { float e = __expf(a[j] - m); w[j] = e; ss += e; }
    float inv = 1.0f / ss;
    for (int j = 0; j < 32; ++j) w[j] *= inv;
  }
  __syncthreads();
  int h = tid;
  float m = 0.0f;
  for (int j = 0; j < 32; ++j) m += w[j] * hist_hidden[j * HID + h];
  moving[(size_t)i * HID + h] = m;
  mv[h] = m;
  __syncthreads();
  if (wid == 0) {
    float sv = 0.0f;
#pragma unroll
    for (int j = 0; j < 8; ++j) { int hh = lane + 32 * j; sv += mv[hh] * score_w[hh]; }
#pragma unroll
    for (int o = 16; o > 0; o >>= 1) sv += __shfl_xor(sv, o, 32);
    if (lane == 0) scores[i] = sv;
  }
}

// ---------------------- row softmax over 128 candidates per s
__global__ void softmax_rows_kernel(const float* __restrict__ scores, float* __restrict__ probs) {
  int s = blockIdx.x, tid = threadIdx.x;  // 128 threads
  __shared__ float red[128];
  float v = scores[s * 128 + tid];
  red[tid] = v; __syncthreads();
  for (int o = 64; o > 0; o >>= 1) { if (tid < o) red[tid] = fmaxf(red[tid], red[tid + o]); __syncthreads(); }
  float m = red[0]; __syncthreads();
  float e = __expf(v - m);
  red[tid] = e; __syncthreads();
  for (int o = 64; o > 0; o >>= 1) { if (tid < o) red[tid] += red[tid + o]; __syncthreads(); }
  probs[s * 128 + tid] = e / red[0];
}

// =============================================================== host side
extern "C" void kernel_launch(void* const* d_in, const int* in_sizes, int n_in,
                              void* d_out, int out_size, void* d_ws, size_t ws_size,
                              hipStream_t stream) {
  (void)in_sizes; (void)n_in; (void)out_size; (void)ws_size;
  const int*   uid_p     = (const int*)d_in[0];
  const int*   cur_trace = (const int*)d_in[1];
  const int*   hist_trc  = (const int*)d_in[2];
  const int*   cand_set  = (const int*)d_in[3];
  const float* uid_emb   = (const float*)d_in[4];
  const float* loc_emb   = (const float*)d_in[5];
  const float* wd_emb    = (const float*)d_in[6];
  const float* hr_emb    = (const float*)d_in[7];
  const float* W_ih      = (const float*)d_in[8];
  const float* W_hh      = (const float*)d_in[9];
  const float* b_ih      = (const float*)d_in[10];
  const float* b_hh      = (const float*)d_in[11];
  const float* intra_w1  = (const float*)d_in[12];
  const float* intra_w2  = (const float*)d_in[13];
  const float* intra_out = (const float*)d_in[14];
  const float* inter_w1  = (const float*)d_in[15];
  const float* inter_w2  = (const float*)d_in[16];
  const float* inter_out = (const float*)d_in[17];
  const float* score_w   = (const float*)d_in[18];
  float* out_probs  = (float*)d_out;
  float* out_moving = out_probs + 32 * 128;

  // ---- workspace carve (~33 MB) ----
  char* wsb = (char*)d_ws;
  size_t off = 0;
  auto carve = [&](size_t bytes) -> char* {
    off = (off + 255) & ~(size_t)255;
    char* p = wsb + off; off += bytes; return p;
  };
  float*    XP        = (float*)carve((size_t)4096 * G3 * 4);     // x-proj buffer (reused later)
  _Float16* A16       = (_Float16*)carve((size_t)4096 * EMBD * 2);
  _Float16* Wih16     = (_Float16*)carve((size_t)G3 * EMBD * 2);
  _Float16* Whh16     = (_Float16*)carve((size_t)G3 * HID * 2);
  _Float16* iw1_16    = (_Float16*)carve((size_t)HID * HID * 2);
  _Float16* iw2_16    = (_Float16*)carve((size_t)HID * HID * 2);
  _Float16* ew1_16    = (_Float16*)carve((size_t)HID * HID * 2);
  _Float16* ew2_16    = (_Float16*)carve((size_t)HID * HID * 2);
  float*    gh        = (float*)carve((size_t)32 * G3 * 4);
  float*    hist_hs   = (float*)carve((size_t)32 * 64 * HID * 4);
  _Float16* hist_hs16 = (_Float16*)carve((size_t)32 * 64 * HID * 2);
  float*    cur_hs    = (float*)carve((size_t)32 * 32 * HID * 4);
  _Float16* cur_hs16  = (_Float16*)carve((size_t)32 * 32 * HID * 2);
  float*    cand_h    = (float*)carve((size_t)4096 * HID * 4);
  _Float16* cand_h16  = (_Float16*)carve((size_t)4096 * HID * 2);
  float*    q_hist    = (float*)carve((size_t)32 * HID * 4);
  float*    hist_ctx  = (float*)carve((size_t)32 * HID * 4);
  _Float16* hist_ctx16= (_Float16*)carve((size_t)32 * HID * 2);
  float*    cand_ctx  = (float*)carve((size_t)4096 * HID * 4);
  _Float16* cand_ctx16= (_Float16*)carve((size_t)4096 * HID * 2);
  float*    kh2       = (float*)carve((size_t)32 * HID * 4);
  float*    scores    = (float*)carve((size_t)4096 * 4);
  // alias into XP once the candidate x-proj has been consumed:
  float* q_cand = XP;                                  // 4096*256
  float* k_last = XP + (size_t)4096 * HID;             // 4096*256
  float* k_cur  = XP + (size_t)2 * 4096 * HID;         // 1024*256
  float* k_hist = XP + (size_t)2 * 4096 * HID + 1024 * HID; // 2048*256
  float* qh2    = q_cand;                              // reused after cand_attn

  auto cdiv = [](int a, int b) { return (a + b - 1) / b; };
  auto gemm = [&](const _Float16* A, long lda, const _Float16* W, const float* bias, int act,
                  float* D, long ldd, int M, int N, int K) {
    int waves = (M / 16) * (N / 16);
    gemm_xwT_kernel<<<cdiv(waves, 8), 256, 0, stream>>>(A, lda, W, bias, act, D, ldd, M, N, K);
  };

  // weight conversions to f16
  cvt_f16_kernel<<<cdiv(G3 * EMBD, 256), 256, 0, stream>>>(W_ih, Wih16, G3 * EMBD);
  cvt_f16_kernel<<<cdiv(G3 * HID, 256), 256, 0, stream>>>(W_hh, Whh16, G3 * HID);
  cvt_f16_kernel<<<cdiv(HID * HID, 256), 256, 0, stream>>>(intra_w1, iw1_16, HID * HID);
  cvt_f16_kernel<<<cdiv(HID * HID, 256), 256, 0, stream>>>(intra_w2, iw2_16, HID * HID);
  cvt_f16_kernel<<<cdiv(HID * HID, 256), 256, 0, stream>>>(inter_w1, ew1_16, HID * HID);
  cvt_f16_kernel<<<cdiv(HID * HID, 256), 256, 0, stream>>>(inter_w2, ew2_16, HID * HID);

  // ---- history GRU (B=32, T=64) ----
  {
    int tot = 32 * 64 * EMBD;
    gather_trace_kernel<<<cdiv(tot, 256), 256, 0, stream>>>(hist_trc, uid_p, uid_emb, loc_emb,
                                                            wd_emb, hr_emb, A16, tot);
    gemm(A16, EMBD, Wih16, b_ih, 0, XP, G3, 2048, G3, EMBD);
    gru_seq_kernel<<<1, 1024, 0, stream>>>(XP, Whh16, b_hh, gh, hist_hs, hist_hs16, 64);
  }
  // ---- current-trace GRU (B=32, T=32) ----
  {
    int tot = 32 * 32 * EMBD;
    gather_trace_kernel<<<cdiv(tot, 256), 256, 0, stream>>>(cur_trace, uid_p, uid_emb, loc_emb,
                                                            wd_emb, hr_emb, A16, tot);
    gemm(A16, EMBD, Wih16, b_ih, 0, XP, G3, 1024, G3, EMBD);
    gru_seq_kernel<<<1, 1024, 0, stream>>>(XP, Whh16, b_hh, gh, cur_hs, cur_hs16, 32);
  }
  // ---- candidate final GRU step (shared prefix: gh depends only on s) ----
  {
    int tot = 4096 * EMBD;
    gather_cand_kernel<<<cdiv(tot, 256), 256, 0, stream>>>(cand_set, cur_trace, uid_p, uid_emb,
                                                           loc_emb, wd_emb, hr_emb, A16, tot);
    gemm(A16, EMBD, Wih16, b_ih, 0, XP, G3, 4096, G3, EMBD);
    gemm(cur_hs16 + 31 * HID, 32 * HID, Whh16, b_hh, 0, gh, G3, 32, G3, HID);
    cand_gru_combine_kernel<<<cdiv(4096 * HID, 256), 256, 0, stream>>>(XP, gh, cur_hs, cand_h, cand_h16);
  }
  // ---- history intra attention (XP region now free for reuse) ----
  gemm(hist_hs16 + 63 * HID, 64 * HID, iw1_16, (const float*)nullptr, 1, q_hist, HID, 32, HID, HID);
  gemm(hist_hs16, HID, iw2_16, (const float*)nullptr, 1, k_hist, HID, 2048, HID, HID);
  attn_ctx_kernel<<<32, 256, 0, stream>>>(q_hist, k_hist, hist_hs, intra_out, hist_ctx, hist_ctx16, 64);
  // ---- candidate intra attention ----
  gemm(cand_h16, HID, iw1_16, (const float*)nullptr, 1, q_cand, HID, 4096, HID, HID);
  gemm(cand_h16, HID, iw2_16, (const float*)nullptr, 1, k_last, HID, 4096, HID, HID);
  gemm(cur_hs16, HID, iw2_16, (const float*)nullptr, 1, k_cur, HID, 1024, HID, HID);
  cand_attn_kernel<<<4096, 256, 0, stream>>>(q_cand, k_cur, k_last, cur_hs, cand_h, intra_out,
                                             cand_ctx, cand_ctx16);
  // ---- inter attention + outputs ----
  gemm(cand_ctx16, HID, ew1_16, (const float*)nullptr, 1, qh2, HID, 4096, HID, HID);
  gemm(hist_ctx16, HID, ew2_16, (const float*)nullptr, 1, kh2, HID, 32, HID, HID);
  inter_attn_kernel<<<4096, 256, 0, stream>>>(qh2, kh2, inter_out, hist_ctx, score_w,
                                              out_moving, scores);
  softmax_rows_kernel<<<32, 128, 0, stream>>>(scores, out_probs);
}
